// LSTMv2_46651934769188
// MI455X (gfx1250) — matrix operand order repaired and checked by reading, not compiled, not run on hardware
//
#include <hip/hip_runtime.h>
#include <hip/hip_bf16.h>

// ---------------------------------------------------------------------------
// 2-layer LSTM (B=256, T=512, IN=19, H=512) + MLP head.
// Fused persistent scan, software-pipelined across layers: phase t computes
// layer1's h1[t] AND layer2's h2[t-1] (both depend only on phase t-1 data),
// so only ONE grid barrier per phase (513 total instead of 1024).
// Recurrent GEMMs via v_wmma_f32_16x16x32_bf16; weights resident in LDS.
// ---------------------------------------------------------------------------

typedef __bf16 bf16_t;
typedef __attribute__((ext_vector_type(16))) __bf16 v16bf;
typedef __attribute__((ext_vector_type(8)))  __bf16 v8bf;
typedef __attribute__((ext_vector_type(8)))  float  v8f;

#define Hdim     512
#define Bdim     256
#define Tdim     512
#define INdim    19
#define NWG      32       // workgroups; each owns 16 columns of H
#define HS       16       // H columns per workgroup
#define NTHREADS 512      // 16 waves of 32

union ABfrag { v16bf v; v8bf h[2]; };

// A-matrix (16x32 bf16) fragment: lane<16 -> M=lane, K={k0..k0+7, k0+16..k0+23}
//                                 lane>=16 -> M=lane-16, K offset +8.
__device__ __forceinline__ v16bf load_a_frag(const bf16_t* __restrict__ base,
                                             int ldk, int m, int k0, int lane) {
  const int koff = (lane >> 4) << 3;
  const bf16_t* p = base + m * ldk + k0 + koff;
  ABfrag u;
  u.h[0] = *(const v8bf*)(p);
  u.h[1] = *(const v8bf*)(p + 16);
  return u.v;
}

// B-matrix (32x16 bf16) fragment, W stored as B^T row-major (N rows x K cols):
// lane<16 -> N=lane, K={k0..k0+7, k0+16..k0+23} ; lane>=16 -> N=lane-16, K offset +8.
__device__ __forceinline__ v16bf load_b_frag(const bf16_t* __restrict__ base,
                                             int ldk, int n, int k0, int lane) {
  const int koff = (lane >> 4) << 3;
  const bf16_t* p = base + n * ldk + k0 + koff;
  ABfrag u;
  u.h[0] = *(const v8bf*)(p);
  u.h[1] = *(const v8bf*)(p + 16);
  return u.v;
}

__device__ __forceinline__ float sigmoidf_(float x) {
  return 1.0f / (1.0f + __expf(-x));
}
__device__ __forceinline__ float tanhf_(float x) {
  const float xc = fminf(fmaxf(x, -15.0f), 15.0f);
  const float e  = __expf(2.0f * xc);
  return (e - 1.0f) / (e + 1.0f);
}

// Monotonic-counter grid barrier (all NWG workgroups resident).
__device__ __forceinline__ void grid_barrier(unsigned* bar, unsigned& gen) {
  __syncthreads();
  gen += NWG;
  if (threadIdx.x == 0) {
    __threadfence();                       // release h/c writes device-wide
    atomicAdd(bar, 1u);
    while (__hip_atomic_load(bar, __ATOMIC_RELAXED, __HIP_MEMORY_SCOPE_AGENT) < gen) {
      __builtin_amdgcn_s_sleep(2);
    }
    __threadfence();                       // acquire other WGs' writes
  }
  __syncthreads();
}

__global__ void __launch_bounds__(NTHREADS, 1)
lstm_scan_kernel(const float* __restrict__ x,
                 const float* __restrict__ l1Wx, const float* __restrict__ l1bx,
                 const float* __restrict__ l1Wh, const float* __restrict__ l1bh,
                 const float* __restrict__ l2Wx, const float* __restrict__ l2bx,
                 const float* __restrict__ l2Wh, const float* __restrict__ l2bh,
                 bf16_t* __restrict__ h1buf, bf16_t* __restrict__ h2buf,
                 float* __restrict__ c1, float* __restrict__ c2,
                 unsigned* bar)
{
  extern __shared__ char smem_raw[];
  bf16_t* w1h   = (bf16_t*)smem_raw;           // [64][512] bf16  = 64 KB
  bf16_t* w2x   = w1h + 64 * Hdim;             // [64][512] bf16  = 64 KB
  bf16_t* w2h   = w2x + 64 * Hdim;             // [64][512] bf16  = 64 KB
  bf16_t* w1x   = w2h + 64 * Hdim;             // [64][32]  bf16  =  4 KB
  bf16_t* xst   = w1x + 64 * 32;               // [256][32] bf16  = 16 KB
  float*  bias1 = (float*)(xst + Bdim * 32);   // [64]
  float*  bias2 = bias1 + 64;                  // [64]   total ~212.5 KB LDS

  const int wg   = blockIdx.x;
  const int tid  = threadIdx.x;
  const int lane = tid & 31;
  const int wave = tid >> 5;

  // ---- One-time: stage this WG's weight slices (gate rows g*512 + wg*16 + j)
  //      into LDS, converting f32 -> bf16. Row r = g*16 + j, r in [0,64).
  for (int idx = tid; idx < 64 * Hdim; idx += NTHREADS) {
    const int r = idx >> 9, k = idx & (Hdim - 1);
    const int gr = (r >> 4) * Hdim + wg * HS + (r & 15);
    w1h[idx] = (bf16_t)l1Wh[gr * Hdim + k];
    w2x[idx] = (bf16_t)l2Wx[gr * Hdim + k];
    w2h[idx] = (bf16_t)l2Wh[gr * Hdim + k];
  }
  for (int idx = tid; idx < 64 * 32; idx += NTHREADS) {
    const int r = idx >> 5, k = idx & 31;
    const int gr = (r >> 4) * Hdim + wg * HS + (r & 15);
    w1x[idx] = (k < INdim) ? (bf16_t)l1Wx[gr * INdim + k] : (bf16_t)0.0f;
  }
  if (tid < 64) {
    const int gr = (tid >> 4) * Hdim + wg * HS + (tid & 15);
    bias1[tid] = l1bx[gr] + l1bh[gr];
    bias2[tid] = l2bx[gr] + l2bh[gr];
  }
  __syncthreads();

  const int mbase = wave * 16;          // this wave's M tile (rows of B)
  const int mfrag = mbase + (lane & 15);
  const int ncol  = lane & 15;          // N column within a 16-wide gate tile
  const int hcol  = wg * HS + ncol;     // owned H column
  const int mhalf = (lane >> 4) << 3;   // C/D layout: rows +8 for hi lanes
  unsigned  gen   = 0;

  // Pipelined phases: phase t does layer1 step t (t<T) and layer2 step t-1 (t>0).
  for (int t = 0; t <= Tdim; ++t) {
    const int p = t & 1;
    const bf16_t* h1_prev = h1buf + p * (Bdim * Hdim);        // h1[t-1]
    bf16_t*       h1_next = h1buf + (1 - p) * (Bdim * Hdim);  // h1[t]
    const bf16_t* h2_prev = h2buf + p * (Bdim * Hdim);        // h2[t-2]
    bf16_t*       h2_next = h2buf + (1 - p) * (Bdim * Hdim);  // h2[t-1]

    if (t < Tdim) {
      // Stage x_t (256 x 19 f32) -> LDS bf16 padded to 256 x 32.
      for (int idx = tid; idx < Bdim * 32; idx += NTHREADS) {
        const int b = idx >> 5, k = idx & 31;
        xst[idx] = (k < INdim) ? (bf16_t)x[(b * Tdim + t) * INdim + k] : (bf16_t)0.0f;
        if (k == 0 && t + 1 < Tdim)
          __builtin_prefetch(&x[(b * Tdim + t + 1) * INdim], 0, 1);  // global_prefetch_b8
      }
      __syncthreads();

      // ============ Layer 1, step t: gates = x_t@Wx^T + h1[t-1]@Wh^T + b ====
      v8f acc[4];
#pragma unroll
      for (int g = 0; g < 4; ++g) {
        const float bv = bias1[g * 16 + ncol];
#pragma unroll
        for (int j = 0; j < 8; ++j) acc[g][j] = bv;
      }
      {  // K-step 0: padded input projection (A from LDS x-stage)
        const v16bf a = load_a_frag(xst, 32, mfrag, 0, lane);
#pragma unroll
        for (int g = 0; g < 4; ++g) {
          const v16bf b = load_b_frag(w1x, 32, g * 16 + ncol, 0, lane);
          acc[g] = __builtin_amdgcn_wmma_f32_16x16x32_bf16(
              false, a, false, b, (short)0, acc[g], false, false);
        }
      }
#pragma unroll 2
      for (int ks = 0; ks < Hdim / 32; ++ks) {  // recurrent part, K = 512
        const v16bf a = load_a_frag(h1_prev, Hdim, mfrag, ks * 32, lane);
#pragma unroll
        for (int g = 0; g < 4; ++g) {
          const v16bf b = load_b_frag(w1h, Hdim, g * 16 + ncol, ks * 32, lane);
          acc[g] = __builtin_amdgcn_wmma_f32_16x16x32_bf16(
              false, a, false, b, (short)0, acc[g], false, false);
        }
      }
#pragma unroll
      for (int j = 0; j < 8; ++j) {  // gate nonlinearity + state update
        const int m = mbase + mhalf + j;
        const float iv = sigmoidf_(acc[0][j]);
        const float fv = sigmoidf_(acc[1][j]);
        const float gv = tanhf_(acc[2][j]);
        const float ov = sigmoidf_(acc[3][j]);
        const float cn = fv * c1[m * Hdim + hcol] + iv * gv;
        c1[m * Hdim + hcol] = cn;
        h1_next[m * Hdim + hcol] = (bf16_t)(ov * tanhf_(cn));
      }
    }

    if (t > 0) {
      // ==== Layer 2, step t-1: gates = h1[t-1]@Wx^T + h2[t-2]@Wh^T + b ======
      v8f acc[4];
#pragma unroll
      for (int g = 0; g < 4; ++g) {
        const float bv = bias2[g * 16 + ncol];
#pragma unroll
        for (int j = 0; j < 8; ++j) acc[g][j] = bv;
      }
#pragma unroll 2
      for (int ks = 0; ks < Hdim / 32; ++ks) {  // input projection from h1[t-1]
        const v16bf a = load_a_frag(h1_prev, Hdim, mfrag, ks * 32, lane);
#pragma unroll
        for (int g = 0; g < 4; ++g) {
          const v16bf b = load_b_frag(w2x, Hdim, g * 16 + ncol, ks * 32, lane);
          acc[g] = __builtin_amdgcn_wmma_f32_16x16x32_bf16(
              false, a, false, b, (short)0, acc[g], false, false);
        }
      }
#pragma unroll 2
      for (int ks = 0; ks < Hdim / 32; ++ks) {  // recurrent part from h2[t-2]
        const v16bf a = load_a_frag(h2_prev, Hdim, mfrag, ks * 32, lane);
#pragma unroll
        for (int g = 0; g < 4; ++g) {
          const v16bf b = load_b_frag(w2h, Hdim, g * 16 + ncol, ks * 32, lane);
          acc[g] = __builtin_amdgcn_wmma_f32_16x16x32_bf16(
              false, a, false, b, (short)0, acc[g], false, false);
        }
      }
#pragma unroll
      for (int j = 0; j < 8; ++j) {
        const int m = mbase + mhalf + j;
        const float iv = sigmoidf_(acc[0][j]);
        const float fv = sigmoidf_(acc[1][j]);
        const float gv = tanhf_(acc[2][j]);
        const float ov = sigmoidf_(acc[3][j]);
        const float cn = fv * c2[m * Hdim + hcol] + iv * gv;
        c2[m * Hdim + hcol] = cn;
        h2_next[m * Hdim + hcol] = (bf16_t)(ov * tanhf_(cn));
      }
    }

    grid_barrier(bar, gen);   // single device-wide sync per phase
  }
}

// ------------------------------- MLP head ----------------------------------
__global__ void fc1_kernel(const bf16_t* __restrict__ h, const float* __restrict__ W,
                           const float* __restrict__ bias, float* __restrict__ y) {
  const int idx = blockIdx.x * blockDim.x + threadIdx.x;   // 256*256
  const int row = idx >> 8, col = idx & 255;
  const bf16_t* hr = h + row * Hdim;
  const float*  wr = W + col * Hdim;
  float s = bias[col];
  for (int k = 0; k < Hdim; ++k) s += (float)hr[k] * wr[k];
  y[idx] = fmaxf(s, 0.0f);
}

__global__ void fc2_kernel(const float* __restrict__ y1, const float* __restrict__ W,
                           const float* __restrict__ bias, float* __restrict__ y2) {
  const int idx = blockIdx.x * blockDim.x + threadIdx.x;   // 256*128
  const int row = idx >> 7, col = idx & 127;
  const float* xr = y1 + row * 256;
  const float* wr = W + col * 256;
  float s = bias[col];
  for (int k = 0; k < 256; ++k) s += xr[k] * wr[k];
  y2[idx] = fmaxf(s, 0.0f);
}

__global__ void fc3_kernel(const float* __restrict__ y2, const float* __restrict__ W,
                           const float* __restrict__ bias, float* __restrict__ out) {
  const int row = blockIdx.x * blockDim.x + threadIdx.x;   // 256
  if (row < Bdim) {
    const float* xr = y2 + row * 128;
    float s = bias[0];
    for (int k = 0; k < 128; ++k) s += xr[k] * W[k];
    out[row] = s;
  }
}

extern "C" void kernel_launch(void* const* d_in, const int* in_sizes, int n_in,
                              void* d_out, int out_size, void* d_ws, size_t ws_size,
                              hipStream_t stream) {
  (void)in_sizes; (void)n_in; (void)out_size; (void)ws_size;
  const float* x    = (const float*)d_in[0];
  const float* l1Wx = (const float*)d_in[1];
  const float* l1bx = (const float*)d_in[2];
  const float* l1Wh = (const float*)d_in[3];
  const float* l1bh = (const float*)d_in[4];
  const float* l2Wx = (const float*)d_in[5];
  const float* l2bx = (const float*)d_in[6];
  const float* l2Wh = (const float*)d_in[7];
  const float* l2bh = (const float*)d_in[8];
  const float* fc1W = (const float*)d_in[9];
  const float* fc1b = (const float*)d_in[10];
  const float* fc2W = (const float*)d_in[11];
  const float* fc2b = (const float*)d_in[12];
  const float* fc3W = (const float*)d_in[13];
  const float* fc3b = (const float*)d_in[14];

  char* ws = (char*)d_ws;
  bf16_t*   h1  = (bf16_t*)(ws);                       // 2 x 256x512 bf16 (512 KB)
  bf16_t*   h2  = (bf16_t*)(ws + (512u << 10));        // 2 x 256x512 bf16 (512 KB)
  float*    c1  = (float*) (ws + (1024u << 10));       // 256x512 f32 (512 KB)
  float*    c2  = (float*) (ws + (1536u << 10));       // 256x512 f32 (512 KB)
  unsigned* bar = (unsigned*)(ws + (2048u << 10));     // barrier counter
  float*    y1  = (float*) (ws + (2048u << 10) + 256); // 256x256 f32
  float*    y2  = y1 + 256 * 256;                      // 256x128 f32

  // Zero h/c state + barrier counter every launch (graph-capture safe).
  hipMemsetAsync(d_ws, 0, (2048u << 10) + 256, stream);

  const size_t smem = (size_t)(3 * 64 * Hdim + 64 * 32 + Bdim * 32) * sizeof(bf16_t)
                    + 128 * sizeof(float);
  hipFuncSetAttribute((const void*)lstm_scan_kernel,
                      hipFuncAttributeMaxDynamicSharedMemorySize, (int)smem);

  lstm_scan_kernel<<<NWG, NTHREADS, smem, stream>>>(
      x, l1Wx, l1bx, l1Wh, l1bh, l2Wx, l2bx, l2Wh, l2bh,
      h1, h2, c1, c2, bar);

  // Final phase t=T=512 (p=0) wrote h2[T-1] into buffer (1-p) = 1.
  const bf16_t* hfin = h2 + Bdim * Hdim;
  fc1_kernel<<<256, 256, 0, stream>>>(hfin, fc1W, fc1b, y1);
  fc2_kernel<<<128, 256, 0, stream>>>(y1, fc2W, fc2b, y2);
  fc3_kernel<<<1, 256, 0, stream>>>(y2, fc3W, fc3b, (float*)d_out);
}
